// xLSTMModel_68384469287110
// MI455X (gfx1250) — matrix operand order repaired and checked by reading, not compiled, not run on hardware
//
#include <hip/hip_runtime.h>
#include <hip/hip_bf16.h>
#include <math.h>

// ---------------- model constants ----------------
#define IN_DIM 64
#define HID    512
#define NBLOCKS 4
#define OUT_DIM 8
#define CTX    512
#define HEADS  4
#define HD     128          // HID / HEADS
#define CONVK  4
#define UP     665          // int(1.3*512)
#define UPP    672          // padded leading dim (16B aligned)
#define BATCH  32
#define ROWS   (BATCH*CTX) // 16384

typedef __attribute__((ext_vector_type(16))) _Float16 v16h;
typedef __attribute__((ext_vector_type(8)))  _Float16 h8;
typedef __attribute__((ext_vector_type(4)))  _Float16 h4;
typedef __attribute__((ext_vector_type(8)))  float    v8f;

// ---------------- WMMA tiled GEMM ----------------
// C[M,N] = act(A[M,K] * B[K,N] + bias), optionally accumulated into C.
// f32 in/out, f16 math, f32 accumulate (v_wmma_f32_16x16x32_f16).
// - LDS tiles stored as f16 pre-arranged for the WMMA lane layout
//   (A and B^T fragments: one 16-half run each) so each
//   fragment is built from ds_load_b128s with no converts.
// - Double-buffered LDS: stage chunk k+1 while computing chunk k;
//   one barrier per iteration.
// - Each wave owns two independent 16x16 accumulators (shares the A
//   fragment) -> two back-to-back independent v_wmma per iteration.
#define BM 64
#define BN 64
#define BK 32
#define LPAD 8              // pad rows to 40 halves = 80B (16B-aligned rows)
#define ACT_NONE 0
#define ACT_GELU 1

__global__ __launch_bounds__(256) void gemm_wmma_k(
    const float* __restrict__ A, int lda, long aOffZ,
    const float* __restrict__ Bw, int ldb, long bOffZ,
    float* __restrict__ C, int ldc, long cOffZ,
    const float* __restrict__ bias, long biasOffZ,
    int M, int N, int K, int act, int accumulate)
{
    const int z = blockIdx.z;
    A  += (long)z * aOffZ;
    Bw += (long)z * bOffZ;
    C  += (long)z * cOffZ;
    if (bias) bias += (long)z * biasOffZ;

    __shared__ __attribute__((aligned(16))) _Float16 As_h[2][BM][BK + LPAD];
    __shared__ __attribute__((aligned(16))) _Float16 BsT[2][BN][BK + LPAD];

    const int t    = threadIdx.x;
    const int lane = t & 31;          // wave32
    const int wave = t >> 5;          // 8 waves
    const int wm   = wave & 3;        // 4 row tiles of 16
    const int wn   = wave >> 2;       // 2 col tiles of 32 (2x16 each)
    const int rowBase = blockIdx.y * BM;
    const int colBase = blockIdx.x * BN;

    const int mloc = lane & 15;
    const int half = lane >> 4;

    auto stageA = [&](int buf, int k0) {
        const bool aVec = ((lda & 3) == 0) && (k0 + BK <= K);
        if (aVec) {
            #pragma unroll
            for (int i = 0; i < 2; ++i) {
                int q = t + i * 256;
                int r = q >> 3, c4 = (q & 7) * 4;
                int gr = rowBase + r;
                int grc = (gr < M) ? gr : (M - 1);
                const float4 v = *(const float4*)(A + (long)grc * lda + (k0 + c4));
                float msk = (gr < M) ? 1.f : 0.f;     // select, not branch
                h4 hv;
                hv[0] = (_Float16)(v.x * msk); hv[1] = (_Float16)(v.y * msk);
                hv[2] = (_Float16)(v.z * msk); hv[3] = (_Float16)(v.w * msk);
                *(h4*)&As_h[buf][r][c4] = hv;
            }
        } else {
            #pragma unroll
            for (int i = 0; i < 8; ++i) {
                int li = t + i * 256;
                int r = li >> 5, c = li & 31;
                int gr = rowBase + r, gc = k0 + c;
                long idx = (long)((gr < M) ? gr : M - 1) * lda + ((gc < K) ? gc : K - 1);
                float v = A[idx];                      // always-valid address
                v = (gr < M && gc < K) ? v : 0.f;      // cndmask select
                As_h[buf][r][c] = (_Float16)v;
            }
        }
    };
    auto stageB = [&](int buf, int k0) {
        const bool bVec = ((ldb & 3) == 0) && (k0 + BK <= K) && (colBase + BN <= N);
        if (bVec) {
            #pragma unroll
            for (int i = 0; i < 2; ++i) {
                int q = t + i * 256;
                int r = q >> 4, c4 = (q & 15) * 4;
                const float4 v = *(const float4*)(Bw + (long)(k0 + r) * ldb + (colBase + c4));
                BsT[buf][c4 + 0][r] = (_Float16)v.x;
                BsT[buf][c4 + 1][r] = (_Float16)v.y;
                BsT[buf][c4 + 2][r] = (_Float16)v.z;
                BsT[buf][c4 + 3][r] = (_Float16)v.w;
            }
        } else {
            #pragma unroll
            for (int i = 0; i < 8; ++i) {
                int li = t + i * 256;
                int r = li >> 6, c = li & 63;
                int gk = k0 + r, gc = colBase + c;
                long idx = (long)((gk < K) ? gk : K - 1) * ldb + ((gc < N) ? gc : N - 1);
                float v = Bw[idx];
                v = (gk < K && gc < N) ? v : 0.f;
                BsT[buf][c][r] = (_Float16)v;
            }
        }
    };

    v8f acc0 = {};
    v8f acc1 = {};

    const int nk = (K + BK - 1) / BK;
    stageA(0, 0);
    stageB(0, 0);
    __syncthreads();

    for (int it = 0; it < nk; ++it) {
        const int cur = it & 1;
        if (it + 1 < nk) {                 // prefetch next chunk into other buffer
            stageA(cur ^ 1, (it + 1) * BK);
            stageB(cur ^ 1, (it + 1) * BK);
        }

        // ---- A fragment: 16x32 f16, contiguous k order (same as B) ----
        // lane (m = lane%16, half = lane/16):
        //   halves 0..15 <- k = 16*half + idx
        const _Float16* ap = &As_h[cur][wm * 16 + mloc][16 * half];
        h8 alo = *(const h8*)ap;
        h8 ahi = *(const h8*)(ap + 8);
        v16h af = __builtin_shufflevector(alo, ahi,
            0,1,2,3,4,5,6,7,8,9,10,11,12,13,14,15);
        // ---- B fragments (two 16-col subtiles): k = 16*half + idx at fixed n ----
        const _Float16* bp0 = &BsT[cur][wn * 32 + mloc][16 * half];
        h8 b0lo = *(const h8*)bp0;
        h8 b0hi = *(const h8*)(bp0 + 8);
        v16h bf0 = __builtin_shufflevector(b0lo, b0hi,
            0,1,2,3,4,5,6,7,8,9,10,11,12,13,14,15);
        const _Float16* bp1 = &BsT[cur][wn * 32 + 16 + mloc][16 * half];
        h8 b1lo = *(const h8*)bp1;
        h8 b1hi = *(const h8*)(bp1 + 8);
        v16h bf1 = __builtin_shufflevector(b1lo, b1hi,
            0,1,2,3,4,5,6,7,8,9,10,11,12,13,14,15);

        // two independent WMMAs fill the matrix pipe
        acc0 = __builtin_amdgcn_wmma_f32_16x16x32_f16(
            false, af, false, bf0, (short)0, acc0, false, false);
        acc1 = __builtin_amdgcn_wmma_f32_16x16x32_f16(
            false, af, false, bf1, (short)0, acc1, false, false);

        __syncthreads();                   // next-buffer writes visible; cur reusable
    }

    // C/D layout: VGPR r -> M = r + 8*half ; N = lane%16
    const int cn0 = colBase + wn * 32 + mloc;
    const int cn1 = cn0 + 16;
    #pragma unroll
    for (int r = 0; r < 8; ++r) {
        int cm = rowBase + wm * 16 + r + 8 * half;
        if (cm < M) {
            #pragma unroll
            for (int j = 0; j < 2; ++j) {
                int cn = j ? cn1 : cn0;
                if (cn < N) {
                    float v = j ? acc1[r] : acc0[r];
                    if (bias) v += bias[cn];
                    if (act == ACT_GELU) {
                        float x = v;
                        v = 0.5f * x * (1.f + tanhf(0.7978845608028654f *
                                                    (x + 0.044715f * x * x * x)));
                    }
                    long ci = (long)cm * ldc + cn;
                    if (accumulate) C[ci] += v; else C[ci] = v;
                }
            }
        }
    }
}

// ---------------- LayerNorm (row = 512) ----------------
__global__ __launch_bounds__(256) void layernorm_k(
    const float* __restrict__ src, long srcStride,
    float* __restrict__ dst, long dstStride,
    const float* __restrict__ w, const float* __restrict__ b)
{
    const long row = blockIdx.x;
    const float* s = src + row * srcStride;
    float* d = dst + row * dstStride;
    const int t = threadIdx.x;
    float x0 = s[t], x1 = s[t + 256];
    __shared__ float red[256];
    red[t] = x0 + x1; __syncthreads();
    for (int o = 128; o > 0; o >>= 1) { if (t < o) red[t] += red[t + o]; __syncthreads(); }
    float mu = red[0] * (1.f / 512.f);
    __syncthreads();
    float d0 = x0 - mu, d1 = x1 - mu;
    red[t] = d0 * d0 + d1 * d1; __syncthreads();
    for (int o = 128; o > 0; o >>= 1) { if (t < o) red[t] += red[t + o]; __syncthreads(); }
    float rstd = rsqrtf(red[0] * (1.f / 512.f) + 1e-6f);
    d[t]       = d0 * rstd * w[t] + b[t];
    d[t + 256] = d1 * rstd * w[t + 256] + b[t + 256];
}

// ---------------- causal depthwise conv + SiLU ----------------
__global__ void conv_silu_k(const float* __restrict__ x,
                            const float* __restrict__ w,
                            const float* __restrict__ b,
                            float* __restrict__ y, long total)
{
    long i = (long)blockIdx.x * blockDim.x + threadIdx.x;
    if (i >= total) return;
    int d = (int)(i & (HID - 1));
    long bs = i >> 9;                 // /HID
    int s = (int)(bs & (CTX - 1));
    float acc = b[d];
    #pragma unroll
    for (int k = 0; k < CONVK; ++k) {
        int ss = s - (CONVK - 1) + k;
        if (ss >= 0) acc += x[i + (long)(ss - s) * HID] * w[k * HID + d];
    }
    y[i] = acc / (1.f + __expf(-acc));   // silu
}

// ---------------- sLSTM sequential scan ----------------
// One workgroup per (batch, head); 128 threads, one per hd channel.
// Gates g0..g3 already include bias. R layout [4, HEADS, HD, HD].
__global__ __launch_bounds__(128) void slstm_scan_k(
    const float* __restrict__ g0, const float* __restrict__ g1,
    const float* __restrict__ g2, const float* __restrict__ g3,
    const float* __restrict__ R, float* __restrict__ hs)
{
    const int bh = blockIdx.x;
    const int b = bh >> 2, h = bh & 3;
    const int d = threadIdx.x;
    __shared__ float hsh[HD];
    hsh[d] = 0.f;
    float c = 0.f, n = 0.f, m = 0.f;
    const float* R0 = R + ((long)(0 * HEADS + h)) * HD * HD;
    const float* R1 = R + ((long)(1 * HEADS + h)) * HD * HD;
    const float* R2 = R + ((long)(2 * HEADS + h)) * HD * HD;
    const float* R3 = R + ((long)(3 * HEADS + h)) * HD * HD;
    __syncthreads();
    for (int s = 0; s < CTX; ++s) {
        long gi = ((long)(b * CTX + s)) * HID + h * HD + d;
        float ir = g0[gi], fr = g1[gi], zr = g2[gi], orr = g3[gi];
        if (s + 1 < CTX) {   // gfx1250 global_prefetch_b8 for next step's gates
            __builtin_prefetch(&g0[gi + HID], 0, 1);
            __builtin_prefetch(&g1[gi + HID], 0, 1);
            __builtin_prefetch(&g2[gi + HID], 0, 1);
            __builtin_prefetch(&g3[gi + HID], 0, 1);
        }
        #pragma unroll 4
        for (int e = 0; e < HD; ++e) {
            float he = hsh[e];
            ir  = fmaf(he, R0[e * HD + d], ir);
            fr  = fmaf(he, R1[e * HD + d], fr);
            zr  = fmaf(he, R2[e * HD + d], zr);
            orr = fmaf(he, R3[e * HD + d], orr);
        }
        float mn = fmaxf(fr + m, ir);
        float iv = __expf(ir - mn);
        float fv = __expf(fr + m - mn);
        c = fv * c + iv * tanhf(zr);
        n = fv * n + iv;
        float hv = (1.f / (1.f + __expf(-orr))) * c / n;
        m = mn;
        hs[gi] = hv;
        __syncthreads();
        hsh[d] = hv;
        __syncthreads();
    }
}

// ---------------- per-head groupnorm + residual add ----------------
__global__ __launch_bounds__(128) void gnorm_res_k(
    const float* __restrict__ hs, const float* __restrict__ gw,
    float* __restrict__ x)
{
    const long row = blockIdx.x;
    const int t = threadIdx.x;
    __shared__ float red[128];
    for (int h = 0; h < HEADS; ++h) {
        long idx = row * HID + h * HD + t;
        float v = hs[idx];
        red[t] = v; __syncthreads();
        for (int o = 64; o > 0; o >>= 1) { if (t < o) red[t] += red[t + o]; __syncthreads(); }
        float mu = red[0] * (1.f / 128.f);
        __syncthreads();
        float dv = v - mu;
        red[t] = dv * dv; __syncthreads();
        for (int o = 64; o > 0; o >>= 1) { if (t < o) red[t] += red[t + o]; __syncthreads(); }
        float rstd = rsqrtf(red[0] * (1.f / 128.f) + 1e-6f);
        __syncthreads();
        x[idx] += dv * rstd * gw[h * HD + t];
    }
}

// ---------------- elementwise product (gated FFN) ----------------
__global__ void mul_k(float* __restrict__ a, const float* __restrict__ b, long total)
{
    long i = (long)blockIdx.x * blockDim.x + threadIdx.x;
    if (i < total) a[i] *= b[i];
}

// ---------------- host orchestration ----------------
static inline int ceilDiv(int a, int b) { return (a + b - 1) / b; }

extern "C" void kernel_launch(void* const* d_in, const int* in_sizes, int n_in,
                              void* d_out, int out_size, void* d_ws, size_t ws_size,
                              hipStream_t stream)
{
    (void)in_sizes; (void)n_in; (void)out_size; (void)ws_size;
    // Input flattening: setup_inputs() dict insertion order.
    const float* x_in  = (const float*)d_in[0];
    const float* in_w  = (const float*)d_in[1];
    const float* in_b  = (const float*)d_in[2];
    // per block (13 tensors): ln1_w, ln1_b, conv_w, conv_b, W, R, bias, gn_w,
    //                         ln2_w, ln2_b, ff_Wg, ff_Wu, ff_Wd
    const int BB = 3;
    const float* post_w = (const float*)d_in[BB + NBLOCKS * 13 + 0];
    const float* post_b = (const float*)d_in[BB + NBLOCKS * 13 + 1];
    const float* out_w  = (const float*)d_in[BB + NBLOCKS * 13 + 2];
    const float* out_b  = (const float*)d_in[BB + NBLOCKS * 13 + 3];

    // workspace layout (floats)
    const size_t RSZ = (size_t)ROWS * HID;        // 8.4M floats
    float* ws  = (float*)d_ws;
    float* H   = ws;                              // residual stream
    float* A1  = H  + RSZ;                        // ln1 out / ln2 out
    float* A2  = A1 + RSZ;                        // conv out / scan out
    float* G   = A2 + RSZ;                        // 4 gate buffers (reused by FFN)
    float* LAST = G + 4 * RSZ;                    // 32x512 post-LN rows
    float* G4[4] = { G, G + RSZ, G + 2 * RSZ, G + 3 * RSZ };
    float* YG = G;                                // FFN intermediates live in G
    float* YU = G + (size_t)ROWS * UPP;           // padded leading dim

    const dim3 gblk(256);

    // h = x @ in_w + in_b       [16384,64]x[64,512]
    gemm_wmma_k<<<dim3(HID / BN, ROWS / BM, 1), gblk, 0, stream>>>(
        x_in, IN_DIM, 0, in_w, HID, 0, H, HID, 0, in_b, 0,
        ROWS, HID, IN_DIM, ACT_NONE, 0);

    for (int bi = 0; bi < NBLOCKS; ++bi) {
        const int base = BB + bi * 13;
        const float* ln1w  = (const float*)d_in[base + 0];
        const float* ln1b  = (const float*)d_in[base + 1];
        const float* convw = (const float*)d_in[base + 2];
        const float* convb = (const float*)d_in[base + 3];
        const float* Wp    = (const float*)d_in[base + 4];   // [4,H,HD,HD]
        const float* Rp    = (const float*)d_in[base + 5];   // [4,H,HD,HD]
        const float* biasp = (const float*)d_in[base + 6];   // [4,H,HD]
        const float* gnw   = (const float*)d_in[base + 7];
        const float* ln2w  = (const float*)d_in[base + 8];
        const float* ln2b  = (const float*)d_in[base + 9];
        const float* ffWg  = (const float*)d_in[base + 10];  // [512,665]
        const float* ffWu  = (const float*)d_in[base + 11];  // [512,665]
        const float* ffWd  = (const float*)d_in[base + 12];  // [665,512]

        // ln1
        layernorm_k<<<ROWS, 256, 0, stream>>>(H, HID, A1, HID, ln1w, ln1b);
        // conv + silu
        conv_silu_k<<<ceilDiv(ROWS * HID, 256), 256, 0, stream>>>(
            A1, convw, convb, A2, (long)ROWS * HID);

        // head-wise gate projections, bias folded in (g0,g1 from conv branch; g2,g3 from ln1)
        for (int g = 0; g < 4; ++g) {
            const float* src = (g < 2) ? A2 : A1;
            gemm_wmma_k<<<dim3(HD / BN, ROWS / BM, HEADS), gblk, 0, stream>>>(
                src, HID, (long)HD,
                Wp + (size_t)g * HEADS * HD * HD, HD, (long)HD * HD,
                G4[g], HID, (long)HD,
                biasp + (size_t)g * HEADS * HD, (long)HD,
                ROWS, HD, HD, ACT_NONE, 0);
        }

        // sequential sLSTM scan -> A2
        slstm_scan_k<<<BATCH * HEADS, HD, 0, stream>>>(
            G4[0], G4[1], G4[2], G4[3], Rp, A2);

        // per-head groupnorm * gn_w, residual into H
        gnorm_res_k<<<ROWS, HD, 0, stream>>>(A2, gnw, H);

        // ln2
        layernorm_k<<<ROWS, 256, 0, stream>>>(H, HID, A1, HID, ln2w, ln2b);

        // FFN up (gelu gate + linear up); C has padded ldc=UPP
        gemm_wmma_k<<<dim3(ceilDiv(UP, BN), ROWS / BM, 1), gblk, 0, stream>>>(
            A1, HID, 0, ffWg, UP, 0, YG, UPP, 0, nullptr, 0,
            ROWS, UP, HID, ACT_GELU, 0);
        gemm_wmma_k<<<dim3(ceilDiv(UP, BN), ROWS / BM, 1), gblk, 0, stream>>>(
            A1, HID, 0, ffWu, UP, 0, YU, UPP, 0, nullptr, 0,
            ROWS, UP, HID, ACT_NONE, 0);
        mul_k<<<ceilDiv(ROWS * UPP, 256), 256, 0, stream>>>(YG, YU, (long)ROWS * UPP);
        // FFN down (A lda=UPP aligned), accumulate into residual
        gemm_wmma_k<<<dim3(HID / BN, ROWS / BM, 1), gblk, 0, stream>>>(
            YG, UPP, 0, ffWd, HID, 0, H, HID, 0, nullptr, 0,
            ROWS, HID, UP, ACT_NONE, 1);
    }

    // post-LN on last token of each batch
    layernorm_k<<<BATCH, 256, 0, stream>>>(
        H + (size_t)(CTX - 1) * HID, (long)CTX * HID, LAST, HID, post_w, post_b);

    // out = LAST @ out_w + out_b   [32,512]x[512,8]
    gemm_wmma_k<<<dim3(1, 1, 1), gblk, 0, stream>>>(
        LAST, HID, 0, out_w, OUT_DIM, 0, (float*)d_out, OUT_DIM, 0, out_b, 0,
        BATCH, OUT_DIM, HID, ACT_NONE, 0);
}
